// GraphonDecoder_61873298866831
// MI455X (gfx1250) — matrix operand order repaired and checked by reading, not compiled, not run on hardware
//
#include <hip/hip_runtime.h>
#include <stdint.h>

// MI455X / gfx1250 (CDNA5, wave32) implementation.
// Output is 16.8MB (~0.7us @ 23.3TB/s); reference math is ~23 GFLOP -> compute
// bound. The pair-invariant z_g@W1+b1 block is folded into the GEMM as a
// constant "1" feature whose B-row carries cbias (C operand is then the
// inline-0 SRC2 form), and the circular-distance features factor through
// per-node sin/cos tables: 4 x v_wmma_f32_16x16x32_f16 per 16 output logits.
// Node harmonics are staged per block via global_load_async_to_lds_b128.

typedef __attribute__((ext_vector_type(16))) _Float16 v16h;
typedef __attribute__((ext_vector_type(8)))  float    v8f;

#define TWO_PI_F 6.28318530717958647692f

// ---------------------------------------------------------------------------
// Kernel 1: per-node harmonic table T[i] = {cos(2pi s), sin(2pi s),
// cos(4pi s), sin(4pi s)} and cbias[h] = b1[h] + sum_d z[d] * W1[10+d, h].
// ---------------------------------------------------------------------------
__global__ void graphon_precompute_kernel(const float* __restrict__ zg,
                                          const float* __restrict__ s,
                                          const float* __restrict__ W1,
                                          const float* __restrict__ b1,
                                          float* __restrict__ T,      // [N][4]
                                          float* __restrict__ cbias,  // [64]
                                          int N) {
  int t = blockIdx.x * blockDim.x + threadIdx.x;
  if (t < N) {
    float ang = TWO_PI_F * s[t];
    float c1 = cosf(ang);
    float s1 = sinf(ang);
    float c2 = c1 * c1 - s1 * s1;   // cos(2a)
    float s2 = 2.0f * s1 * c1;      // sin(2a)
    float4 v; v.x = c1; v.y = s1; v.z = c2; v.w = s2;
    ((float4*)T)[t] = v;
  }
  if (t < 64) {
    float acc = b1[t];
    #pragma unroll
    for (int d = 0; d < 32; ++d)
      acc = fmaf(zg[d], W1[(10 + d) * 64 + t], acc);
    cbias[t] = acc;
  }
}

// ---------------------------------------------------------------------------
// Kernel 2: one block per unordered 16x16 tile pair (I <= J).
// 8 waves x 4 iterations = 32 wave-tasks = 2 orientations x 16 i-rows.
// Each wave-task: 4 x v_wmma_f32_16x16x32_f16 -> 16 logits.
// ---------------------------------------------------------------------------
__global__ __launch_bounds__(256) void graphon_tile_kernel(
    const float* __restrict__ W1,     // [42,64] row-major
    const float* __restrict__ W2,     // [64]
    const float* __restrict__ b2,     // [1]
    const float* __restrict__ T,      // [N][4] node harmonics
    const float* __restrict__ cbias,  // [64]
    float* __restrict__ out,          // [N,N]
    int N) {
  const int I = blockIdx.x;
  const int J = blockIdx.y;
  if (I > J) return;                       // block-uniform; EXEC stays all-ones

  __shared__ float4 Tlds[2][16];           // harmonics: buf0 = block I, buf1 = block J
  __shared__ float  Pl[2][16][16][20];     // partials [o][ii][channel-lane][m(+pad)]
  __shared__ float  ldsL[2][16][16];       // reduced logits per orientation

  const int tid  = threadIdx.x;
  const int w    = tid >> 5;               // wave id (0..7)
  const int lane = tid & 31;
  const int hw   = lane >> 4;              // halfwave (selects K-halves)
  const int ln   = lane & 15;              // row M / column N within tile

  // --- stage the 32 needed T rows into LDS via the CDNA5 async path ---------
  if (w == 0) {
    const int blk = hw ? J : I;            // lanes 0-15 -> I rows, 16-31 -> J rows
    const float4* gsrc = ((const float4*)T) + (blk * 16 + ln);
    uint32_t ldsoff = (uint32_t)(uintptr_t)&Tlds[hw][ln];
    asm volatile("global_load_async_to_lds_b128 %0, %1, off"
                 :: "v"(ldsoff), "v"(gsrc) : "memory");
    asm volatile("s_wait_asynccnt 0x0" ::: "memory");
  }

  // --- loop-invariant per-lane data: B fragments, W2 -----------------------
  // B rows: K=0..9 -> W1 head rows; K=10 -> cbias (bias-as-feature); else 0.
  float w2v[4];
  v16h bfrag[4];
  #pragma unroll
  for (int t = 0; t < 4; ++t) {
    const int n = 16 * t + ln;             // global h-channel
    w2v[t] = W2[n];
    v16h bf;
    #pragma unroll
    for (int e = 0; e < 16; ++e) {
      // 16-bit B 32x16: element e <-> K = 8*hw + e (e<8), 16 + 8*hw + (e-8) (e>=8)
      const int K = (e < 8) ? (8 * hw + e) : (16 + 8 * hw + (e - 8));
      float val;
      if (K < 10)       val = W1[K * 64 + n];
      else if (K == 10) val = cbias[n];
      else              val = 0.0f;
      bf[e] = (_Float16)val;
    }
    bfrag[t] = bf;
  }
  const float b2v = b2[0];
  const v8f czero = {};                    // folds to inline SRC2 = 0

  __syncthreads();                         // Tlds visible to all waves

  #pragma unroll
  for (int it = 0; it < 4; ++it) {
    const int task = it * 8 + w;           // 0..31
    const int o    = task >> 4;            // orientation
    const int ii   = task & 15;            // i-row within tile

    const float4 Ti = Tlds[o ? 1 : 0][ii];      // wave-uniform (ds_load)
    const float4 Tj = Tlds[o ? 0 : 1][ln];      // per-lane pair column m = ln
    const float cosd = Ti.x * Tj.x + Ti.y * Tj.y;            // cos 2pi(si-sj)
    const float sind = fabsf(Ti.y * Tj.x - Ti.x * Tj.y);     // sin 2pi*circdist

    // --- A fragment (16-bit A 16x32 layout): lane holds row M = ln.
    // halfwave 0: K=0..7  -> [pe_i | pe_j], K=16..23 -> pad
    // halfwave 1: K=8,9   -> [cosd, sind], K=10 -> 1.0 (bias), rest pad
    v16h a;
    if (hw == 0) {
      a[0] = (_Float16)Ti.x; a[1] = (_Float16)Ti.y;
      a[2] = (_Float16)Ti.z; a[3] = (_Float16)Ti.w;
      a[4] = (_Float16)Tj.x; a[5] = (_Float16)Tj.y;
      a[6] = (_Float16)Tj.z; a[7] = (_Float16)Tj.w;
      #pragma unroll
      for (int e = 8; e < 16; ++e) a[e] = (_Float16)0.0f;
    } else {
      a[0] = (_Float16)cosd; a[1] = (_Float16)sind;
      a[2] = (_Float16)1.0f;               // K=10: bias feature
      #pragma unroll
      for (int e = 3; e < 16; ++e) a[e] = (_Float16)0.0f;
    }

    // --- layer 1: 4 WMMAs, C = inline 0 (bias rides in the K=10 row) ---
    v8f acc[4];
    #pragma unroll
    for (int t = 0; t < 4; ++t) {
      acc[t] = __builtin_amdgcn_wmma_f32_16x16x32_f16(
          /*neg_a=*/false, a, /*neg_b=*/false, bfrag[t],
          /*c_mod=*/(short)0, czero, /*reuse_a=*/false, /*reuse_b=*/false);
    }

    // --- fused relu + layer-2 dot per channel-lane; scatter partials -------
    // Lane (hw, ln) holds rows m = 8*hw..8*hw+7, channel group ln.
    float pr[8];
    #pragma unroll
    for (int r = 0; r < 8; ++r) {
      float v = fmaxf(acc[0][r], 0.0f) * w2v[0];
      v = fmaf(fmaxf(acc[1][r], 0.0f), w2v[1], v);
      v = fmaf(fmaxf(acc[2][r], 0.0f), w2v[2], v);
      v = fmaf(fmaxf(acc[3][r], 0.0f), w2v[3], v);
      pr[r] = v;
    }
    float4 pA, pB;
    pA.x = pr[0]; pA.y = pr[1]; pA.z = pr[2]; pA.w = pr[3];
    pB.x = pr[4]; pB.y = pr[5]; pB.z = pr[6]; pB.w = pr[7];
    *(float4*)&Pl[o][ii][ln][8 * hw]     = pA;   // 2 x ds_store_b128 per task
    *(float4*)&Pl[o][ii][ln][8 * hw + 4] = pB;
  }

  __syncthreads();

  // --- phase 1: reduce partials over the 16 channel-lanes ------------------
  {
    const int o  = tid >> 7;               // orientation
    const int r  = tid & 127;
    const int ii = r >> 3;                 // 0..15
    const int m0 = (r & 7) * 2;            // 0,2,...,14 (two m per thread)
    float s0 = 0.0f, s1 = 0.0f;
    #pragma unroll
    for (int q = 0; q < 16; ++q) {
      const float2 v = *(const float2*)&Pl[o][ii][q][m0];
      s0 += v.x; s1 += v.y;
    }
    ldsL[o][ii][m0]     = s0 + b2v;
    ldsL[o][ii][m0 + 1] = s1 + b2v;
  }

  __syncthreads();

  // --- phase 2: symmetrize out = 0.5*(L + L^T), diag = -1e9; mirror writes --
  {
    const int ii = tid >> 4;
    const int m  = tid & 15;
    const float v = 0.5f * (ldsL[0][ii][m] + ldsL[1][m][ii]);
    const int gi = I * 16 + ii;
    const int gj = J * 16 + m;
    const float vo = (gi == gj) ? -1.0e9f : v;
    out[(size_t)gi * N + gj] = vo;
    out[(size_t)gj * N + gi] = vo;   // I==J duplicates write identical values
  }
}

// ---------------------------------------------------------------------------
extern "C" void kernel_launch(void* const* d_in, const int* in_sizes, int n_in,
                              void* d_out, int out_size, void* d_ws, size_t ws_size,
                              hipStream_t stream) {
  const float* zg = (const float*)d_in[0];   // [1,32]
  const float* s  = (const float*)d_in[1];   // [N]
  const float* W1 = (const float*)d_in[2];   // [42,64]
  const float* b1 = (const float*)d_in[3];   // [64]
  const float* W2 = (const float*)d_in[4];   // [64,1]
  const float* b2 = (const float*)d_in[5];   // [1]
  (void)n_in; (void)out_size; (void)ws_size;

  const int N = in_sizes[1];                 // 2048
  float* out   = (float*)d_out;
  float* cbias = (float*)d_ws;               // 64 floats
  float* T     = cbias + 64;                 // N*4 floats

  const int threads = 256;
  const int blocks  = (N + threads - 1) / threads;
  graphon_precompute_kernel<<<blocks, threads, 0, stream>>>(zg, s, W1, b1, T, cbias, N);

  const int NT = N / 16;
  dim3 grid(NT, NT);
  graphon_tile_kernel<<<grid, 256, 0, stream>>>(W1, W2, b2, T, cbias, out, N);
}
